// MultiheadAttention_36679020708363
// MI455X (gfx1250) — matrix operand order repaired and checked by reading, not compiled, not run on hardware
//
#include <hip/hip_runtime.h>
#include <hip/hip_bf16.h>

// Problem dims (fixed by the reference)
#define B_ 2
#define S_ 2048
#define E_ 1024
#define H_ 16
#define D_ 64
#define BS_ (B_ * S_)

typedef __bf16 bf16_t;
typedef bf16_t v16bf __attribute__((ext_vector_type(16)));
typedef float  v8f   __attribute__((ext_vector_type(8)));

// ---------------- gfx1250 async global->LDS copy (guarded) -----------------
#if defined(__has_builtin)
#  if __has_builtin(__builtin_amdgcn_global_load_async_to_lds_b128)
#    define HAVE_ASYNC_LDS 1
#  endif
#endif
#ifndef HAVE_ASYNC_LDS
#  define HAVE_ASYNC_LDS 0
#endif

#if HAVE_ASYNC_LDS
// Probe-derived signature: (addrspace(1) int4* gsrc, addrspace(3) int4* ldst,
//                           imm int offset, imm int cpol)
typedef int v4i_t __attribute__((vector_size(16)));
typedef __attribute__((address_space(1))) v4i_t gv4i_t;
typedef __attribute__((address_space(3))) v4i_t lv4i_t;
// IOFFSET applies to BOTH the global and LDS address (ISA 15.18 async pseudocode),
// so a linear chunk can be staged with one base pointer + immediate offsets.
#define ASYNC_CP_B128(gsrc, ldst, off) \
  __builtin_amdgcn_global_load_async_to_lds_b128((gv4i_t*)(gsrc), (lv4i_t*)(ldst), (off), 0)
#endif

__device__ __forceinline__ void async_wait0() {
#if HAVE_ASYNC_LDS
#  if __has_builtin(__builtin_amdgcn_s_wait_asynccnt)
  __builtin_amdgcn_s_wait_asynccnt(0);
#  else
  asm volatile("s_wait_asynccnt 0" ::: "memory");
#  endif
#endif
}

// ---------------- WMMA fragment helpers (wave32, ISA 7.12.2 layouts) --------

// A 16x32 bf16: lane L holds row m=L&15; K mapping depends on lane half.
__device__ __forceinline__ int ka_of(int h, int v, int e) {
  return ((v < 4) ? 0 : 16) + h * 8 + ((v & 3) << 1) + e;
}
// B 32x16 bf16: same K mapping as A (lane half h: K=8h..8h+7 then 16+8h..16+8h+7).
__device__ __forceinline__ int kb_of(int h, int v, int e) {
  return ((v < 4) ? 0 : 16) + h * 8 + ((v & 3) << 1) + e;
}

// A fragment from row-major [m][k] bf16 LDS tile (row stride ld).
__device__ __forceinline__ v16bf frag_a(const bf16_t* p, int ld, int lane) {
  const int m = lane & 15, h = lane >> 4;
  v16bf a;
#pragma unroll
  for (int v = 0; v < 8; ++v) {
    a[2 * v]     = p[m * ld + ka_of(h, v, 0)];
    a[2 * v + 1] = p[m * ld + ka_of(h, v, 1)];
  }
  return a;
}

// A fragment with on-the-fly f32 -> bf16 conversion (P matrix in phase C).
__device__ __forceinline__ v16bf frag_a_f32(const float* p, int ld, int lane) {
  const int m = lane & 15, h = lane >> 4;
  v16bf a;
#pragma unroll
  for (int v = 0; v < 8; ++v) {
    a[2 * v]     = (bf16_t)p[m * ld + ka_of(h, v, 0)];
    a[2 * v + 1] = (bf16_t)p[m * ld + ka_of(h, v, 1)];
  }
  return a;
}

// B fragment where the source tile is stored n-major: B[k][n] = p[n*ld + k].
__device__ __forceinline__ v16bf frag_b_nmajor(const bf16_t* p, int ld, int lane) {
  const int n = lane & 15, h = lane >> 4;
  v16bf b;
#pragma unroll
  for (int v = 0; v < 8; ++v) {
    b[2 * v]     = p[n * ld + kb_of(h, v, 0)];
    b[2 * v + 1] = p[n * ld + kb_of(h, v, 1)];
  }
  return b;
}

// B fragment where the source tile is stored k-major: B[k][n] = p[k*ld + n].
__device__ __forceinline__ v16bf frag_b_kmajor(const bf16_t* p, int ld, int lane) {
  const int n = lane & 15, h = lane >> 4;
  v16bf b;
#pragma unroll
  for (int v = 0; v < 8; ++v) {
    b[2 * v]     = p[kb_of(h, v, 0) * ld + n];
    b[2 * v + 1] = p[kb_of(h, v, 1) * ld + n];
  }
  return b;
}

__device__ __forceinline__ v8f wmma_bf16(v16bf a, v16bf b, v8f c) {
  return __builtin_amdgcn_wmma_f32_16x16x32_bf16(false, a, false, b, (short)0, c,
                                                 false, false);
}

// ---------------- Kernel 1: fused Q/K/V projections ------------------------
// out = X @ W^T + b, written as bf16 [B,H,S,D]. Tile: 128(M) x 64(N = one head),
// K-step 32, 8 waves -> 4x2 grid of 32x32 wave tiles. Staging converts f32->bf16
// in VALU, so the async engine is not applicable here.
__global__ void __launch_bounds__(256) qkv_proj_kernel(
    const float* __restrict__ q, const float* __restrict__ k,
    const float* __restrict__ v,
    const float* __restrict__ wq_w, const float* __restrict__ wq_b,
    const float* __restrict__ wk_w, const float* __restrict__ wk_b,
    const float* __restrict__ wv_w, const float* __restrict__ wv_b,
    bf16_t* __restrict__ qh, bf16_t* __restrict__ kh, bf16_t* __restrict__ vh) {
  __shared__ __align__(16) bf16_t As[128 * 32];
  __shared__ __align__(16) bf16_t Bs[64 * 32];

  const int t = threadIdx.x, lane = t & 31, w = t >> 5;
  const int wm = w >> 1, wn = w & 1;
  const int z = blockIdx.z;  // 0=Q 1=K 2=V
  const float* X    = (z == 0) ? q    : (z == 1) ? k    : v;
  const float* W    = (z == 0) ? wq_w : (z == 1) ? wk_w : wv_w;
  const float* bias = (z == 0) ? wq_b : (z == 1) ? wk_b : wv_b;
  bf16_t* dst       = (z == 0) ? qh   : (z == 1) ? kh   : vh;

  const int row0 = blockIdx.x * 128;   // flattened [B*S] rows
  const int n0   = blockIdx.y * 64;    // one head of output features

  v8f acc[2][2];
#pragma unroll
  for (int i = 0; i < 2; ++i)
#pragma unroll
    for (int j = 0; j < 2; ++j)
#pragma unroll
      for (int e = 0; e < 8; ++e) acc[i][j][e] = 0.f;

  for (int kt = 0; kt < E_ / 32; ++kt) {
    __syncthreads();
    // Stage A: 128x32 f32 -> bf16 (float4 loads)
#pragma unroll
    for (int i = 0; i < 4; ++i) {
      int g = t + 256 * i;                 // 1024 float4 groups
      int r = g >> 3, c4 = (g & 7) << 2;
      float4 f = *(const float4*)(X + (size_t)(row0 + r) * E_ + kt * 32 + c4);
      bf16_t* d = As + r * 32 + c4;
      d[0] = (bf16_t)f.x; d[1] = (bf16_t)f.y; d[2] = (bf16_t)f.z; d[3] = (bf16_t)f.w;
    }
    // Stage B: 64x32 of W (row-major [n_out][k_in], k contiguous)
#pragma unroll
    for (int i = 0; i < 2; ++i) {
      int g = t + 256 * i;                 // 512 float4 groups
      int r = g >> 3, c4 = (g & 7) << 2;
      float4 f = *(const float4*)(W + (size_t)(n0 + r) * E_ + kt * 32 + c4);
      bf16_t* d = Bs + r * 32 + c4;
      d[0] = (bf16_t)f.x; d[1] = (bf16_t)f.y; d[2] = (bf16_t)f.z; d[3] = (bf16_t)f.w;
    }
    __syncthreads();

    v16bf af[2], bfr[2];
#pragma unroll
    for (int mi = 0; mi < 2; ++mi)
      af[mi] = frag_a(As + (wm * 32 + mi * 16) * 32, 32, lane);
#pragma unroll
    for (int ni = 0; ni < 2; ++ni)
      bfr[ni] = frag_b_nmajor(Bs + (wn * 32 + ni * 16) * 32, 32, lane);
#pragma unroll
    for (int mi = 0; mi < 2; ++mi)
#pragma unroll
      for (int ni = 0; ni < 2; ++ni)
        acc[mi][ni] = wmma_bf16(af[mi], bfr[ni], acc[mi][ni]);
  }

  // Epilogue: bias, convert to bf16, scatter to [B,H,S,D]
#pragma unroll
  for (int mi = 0; mi < 2; ++mi)
#pragma unroll
    for (int ni = 0; ni < 2; ++ni) {
      int n = n0 + wn * 32 + ni * 16 + (lane & 15);
      float bv = bias[n];
      int hsel = n >> 6, d = n & 63;
#pragma unroll
      for (int i = 0; i < 8; ++i) {
        int row = row0 + wm * 32 + mi * 16 + ((lane >> 4) << 3) + i;
        int bb = row >> 11, s = row & (S_ - 1);
        dst[(((size_t)bb * H_ + hsel) * S_ + s) * D_ + d] =
            (bf16_t)(acc[mi][ni][i] + bv);
      }
    }
}

// ---------------- Kernel 2: fused attention per (b, h, 16-row q-tile) ------
// Keeps the whole 16 x 2048 fp32 score block in LDS (128 KB — CDNA5 WGP LDS),
// so the S x S weights tensor is written to HBM exactly once. K/V chunks are
// staged with gfx1250 async global->LDS copies when available.
__global__ void __launch_bounds__(256) attn_kernel(
    const bf16_t* __restrict__ qh, const bf16_t* __restrict__ kh,
    const bf16_t* __restrict__ vh, const float* __restrict__ dist,
    const int* __restrict__ mask, float* __restrict__ attw,
    bf16_t* __restrict__ ctx) {
  __shared__ __align__(16) float  sc[16 * S_];    // 128 KB score/prob tile
  __shared__ __align__(16) bf16_t qtl[16 * D_];   // 2 KB Q tile
  __shared__ __align__(16) bf16_t kvb[128 * D_];  // 16 KB K/V chunk staging

  const int t = threadIdx.x, lane = t & 31, w = t >> 5;
  const int qt = blockIdx.x, hh = blockIdx.y, bb = blockIdx.z;
  const int q0 = qt * 16;
  const size_t headbase = (((size_t)bb * H_) + hh) * S_ * D_;

  // Stage Q tile (16 x 64 bf16 = 2 KB)
#if HAVE_ASYNC_LDS
  if (t < 128)
    ASYNC_CP_B128(qh + headbase + (size_t)q0 * D_ + t * 8, qtl + t * 8, 0);
  async_wait0();
#else
  if (t < 128)
    ((uint4*)qtl)[t] = ((const uint4*)(qh + headbase + (size_t)q0 * D_))[t];
#endif
  __syncthreads();
  v16bf aq0 = frag_a(qtl, D_, lane);        // k (=d) 0..31
  v16bf aq1 = frag_a(qtl + 32, D_, lane);   // k (=d) 32..63

  // ---- Phase A: raw scores Q K^T -> LDS (each wave owns 16 keys/chunk) ----
  for (int kc = 0; kc < S_ / 128; ++kc) {
    __syncthreads();
#if HAVE_ASYNC_LDS
    {
      const bf16_t* gs = kh + headbase + (size_t)kc * 128 * D_ + t * 8;
      bf16_t* ls = kvb + t * 8;
      ASYNC_CP_B128(gs, ls, 0);
      ASYNC_CP_B128(gs, ls, 4096);
      ASYNC_CP_B128(gs, ls, 8192);
      ASYNC_CP_B128(gs, ls, 12288);
      async_wait0();
    }
#else
#pragma unroll
    for (int i = 0; i < 4; ++i) {
      int g = t + 256 * i;  // 1024 uint4 = 128x64 bf16
      ((uint4*)kvb)[g] =
          ((const uint4*)(kh + headbase + (size_t)kc * 128 * D_))[g];
    }
#endif
    if (kc + 1 < S_ / 128)
      __builtin_prefetch(kh + headbase + (size_t)(kc + 1) * 128 * D_ + t * 32, 0, 1);
    __syncthreads();

    const int n0 = w * 16;  // this wave's keys within the chunk
    v16bf b0 = frag_b_nmajor(kvb + n0 * D_, D_, lane);
    v16bf b1 = frag_b_nmajor(kvb + n0 * D_ + 32, D_, lane);
    v8f acc;
#pragma unroll
    for (int e = 0; e < 8; ++e) acc[e] = 0.f;
    acc = wmma_bf16(aq0, b0, acc);
    acc = wmma_bf16(aq1, b1, acc);
#pragma unroll
    for (int i = 0; i < 8; ++i) {
      int m = ((lane >> 4) << 3) + i;
      int n = lane & 15;
      sc[m * S_ + kc * 128 + n0 + n] = acc[i];
    }
  }
  __syncthreads();

  // ---- Phase B: scale + dist bias + mask + softmax; stream weights out ----
  const float scale = 0.125f;  // D^-0.5 = 1/8
#pragma unroll
  for (int rr = 0; rr < 2; ++rr) {
    const int r = w * 2 + rr;         // 8 waves x 2 rows = 16 rows
    float* srow = sc + r * S_;
    const int qrow = q0 + r;
    const float* drow = dist + ((size_t)bb * S_ + qrow) * S_;
    const int* mrow = mask + (size_t)bb * S_;

    float mx = -3.0e38f;
    for (int c = lane; c < S_; c += 32) {
      float vfl = srow[c] * scale + 0.4f * drow[c];
      if (mrow[c] == 0) vfl = -1.0e9f;
      srow[c] = vfl;
      mx = fmaxf(mx, vfl);
    }
#pragma unroll
    for (int off = 16; off; off >>= 1) mx = fmaxf(mx, __shfl_xor(mx, off, 32));

    float sum = 0.f;
    for (int c = lane; c < S_; c += 32) {
      float e = __expf(srow[c] - mx);
      srow[c] = e;
      sum += e;
    }
#pragma unroll
    for (int off = 16; off; off >>= 1) sum += __shfl_xor(sum, off, 32);
    const float inv = 1.0f / sum;

    float* wrow = attw + (((size_t)bb * H_ + hh) * S_ + qrow) * S_;
    for (int c = lane; c < S_; c += 32) {
      float p = srow[c] * inv;
      srow[c] = p;       // keep normalized P in LDS for phase C
      wrow[c] = p;       // the single HBM write of the weights tensor
    }
  }
  __syncthreads();

  // ---- Phase C: context = P V, all 8 waves ---------------------------------
  // Wave w owns d-cols nd0..nd0+15 and K-steps {2*kh2, 2*kh2+1} of each chunk;
  // the two K-halves are reduced through LDS at the end.
  v8f o;
#pragma unroll
  for (int e = 0; e < 8; ++e) o[e] = 0.f;
  const int nd0 = (w & 3) * 16;
  const int kh2 = w >> 2;  // 0 or 1
  for (int kc = 0; kc < S_ / 128; ++kc) {
    __syncthreads();
#if HAVE_ASYNC_LDS
    {
      const bf16_t* gs = vh + headbase + (size_t)kc * 128 * D_ + t * 8;
      bf16_t* ls = kvb + t * 8;
      ASYNC_CP_B128(gs, ls, 0);
      ASYNC_CP_B128(gs, ls, 4096);
      ASYNC_CP_B128(gs, ls, 8192);
      ASYNC_CP_B128(gs, ls, 12288);
      async_wait0();
    }
#else
#pragma unroll
    for (int i = 0; i < 4; ++i) {
      int g = t + 256 * i;
      ((uint4*)kvb)[g] =
          ((const uint4*)(vh + headbase + (size_t)kc * 128 * D_))[g];
    }
#endif
    if (kc + 1 < S_ / 128)
      __builtin_prefetch(vh + headbase + (size_t)(kc + 1) * 128 * D_ + t * 32, 0, 1);
    __syncthreads();
#pragma unroll
    for (int ks2 = 0; ks2 < 2; ++ks2) {  // this wave's 2 of the 4 K-steps
      int ks = kh2 * 2 + ks2;
      v16bf ap  = frag_a_f32(sc + kc * 128 + ks * 32, S_, lane);
      v16bf bvf = frag_b_kmajor(kvb + ks * 32 * D_ + nd0, D_, lane);
      o = wmma_bf16(ap, bvf, o);
    }
  }
  __syncthreads();  // all P reads done; sc is dead -> reuse for reduction
  if (w >= 4) {
#pragma unroll
    for (int i = 0; i < 8; ++i) {
      int m = ((lane >> 4) << 3) + i;
      int n = lane & 15;
      sc[(w & 3) * 256 + m * 16 + n] = o[i];
    }
  }
  __syncthreads();
  if (w < 4) {
#pragma unroll
    for (int i = 0; i < 8; ++i) {
      int m = ((lane >> 4) << 3) + i;
      int n = lane & 15;
      o[i] += sc[w * 256 + m * 16 + n];
      int qrow = q0 + m;
      int d = nd0 + n;
      // context layout [B,S,H,D] == row-major [B*S, E] for the final GEMM
      ctx[(((size_t)bb * S_ + qrow) * H_ + hh) * D_ + d] = (bf16_t)o[i];
    }
  }
}

// ---------------- Kernel 3: output projection ------------------------------
__global__ void __launch_bounds__(256) out_proj_kernel(
    const bf16_t* __restrict__ ctx, const float* __restrict__ wo_w,
    const float* __restrict__ wo_b, float* __restrict__ out) {
  __shared__ __align__(16) bf16_t As[128 * 32];
  __shared__ __align__(16) bf16_t Bs[64 * 32];
  const int t = threadIdx.x, lane = t & 31, w = t >> 5;
  const int wm = w >> 1, wn = w & 1;
  const int row0 = blockIdx.x * 128, n0 = blockIdx.y * 64;

  v8f acc[2][2];
#pragma unroll
  for (int i = 0; i < 2; ++i)
#pragma unroll
    for (int j = 0; j < 2; ++j)
#pragma unroll
      for (int e = 0; e < 8; ++e) acc[i][j][e] = 0.f;

  for (int kt = 0; kt < E_ / 32; ++kt) {
    __syncthreads();
    // A: 128x32 bf16 tile of the context (pure copy -> async when available)
#if HAVE_ASYNC_LDS
#pragma unroll
    for (int i = 0; i < 2; ++i) {
      int g = t + 256 * i;
      int r = g >> 2, c8 = (g & 3) << 3;
      ASYNC_CP_B128(ctx + (size_t)(row0 + r) * E_ + kt * 32 + c8,
                    As + r * 32 + c8, 0);
    }
#else
#pragma unroll
    for (int i = 0; i < 2; ++i) {
      int g = t + 256 * i;
      int r = g >> 2, c8 = (g & 3) << 3;
      *(uint4*)(As + r * 32 + c8) =
          *(const uint4*)(ctx + (size_t)(row0 + r) * E_ + kt * 32 + c8);
    }
#endif
    // B: 64x32 of wo_w, f32 -> bf16 (conversion stays in VALU)
#pragma unroll
    for (int i = 0; i < 2; ++i) {
      int g = t + 256 * i;
      int r = g >> 3, c4 = (g & 7) << 2;
      float4 f = *(const float4*)(wo_w + (size_t)(n0 + r) * E_ + kt * 32 + c4);
      bf16_t* d = Bs + r * 32 + c4;
      d[0] = (bf16_t)f.x; d[1] = (bf16_t)f.y; d[2] = (bf16_t)f.z; d[3] = (bf16_t)f.w;
    }
    async_wait0();
    __syncthreads();

    v16bf af[2], bfr[2];
#pragma unroll
    for (int mi = 0; mi < 2; ++mi)
      af[mi] = frag_a(As + (wm * 32 + mi * 16) * 32, 32, lane);
#pragma unroll
    for (int ni = 0; ni < 2; ++ni)
      bfr[ni] = frag_b_nmajor(Bs + (wn * 32 + ni * 16) * 32, 32, lane);
#pragma unroll
    for (int mi = 0; mi < 2; ++mi)
#pragma unroll
      for (int ni = 0; ni < 2; ++ni)
        acc[mi][ni] = wmma_bf16(af[mi], bfr[ni], acc[mi][ni]);
  }

#pragma unroll
  for (int mi = 0; mi < 2; ++mi)
#pragma unroll
    for (int ni = 0; ni < 2; ++ni) {
      int n = n0 + wn * 32 + ni * 16 + (lane & 15);
      float bv = wo_b[n];
#pragma unroll
      for (int i = 0; i < 8; ++i) {
        int row = row0 + wm * 32 + mi * 16 + ((lane >> 4) << 3) + i;
        out[(size_t)row * E_ + n] = acc[mi][ni][i] + bv;
      }
    }
}

// ---------------- Host launcher --------------------------------------------
extern "C" void kernel_launch(void* const* d_in, const int* in_sizes, int n_in,
                              void* d_out, int out_size, void* d_ws,
                              size_t ws_size, hipStream_t stream) {
  const float* q    = (const float*)d_in[0];
  const float* k    = (const float*)d_in[1];
  const float* v    = (const float*)d_in[2];
  const float* dist = (const float*)d_in[3];
  const int*   mask = (const int*)d_in[4];
  const float* wq_w = (const float*)d_in[5];
  const float* wq_b = (const float*)d_in[6];
  const float* wk_w = (const float*)d_in[7];
  const float* wk_b = (const float*)d_in[8];
  const float* wv_w = (const float*)d_in[9];
  const float* wv_b = (const float*)d_in[10];
  const float* wo_w = (const float*)d_in[11];
  const float* wo_b = (const float*)d_in[12];

  float* out  = (float*)d_out;                       // [B,S,E]
  float* attw = out + (size_t)B_ * S_ * E_;          // [B,H,S,S]

  // Workspace: Q/K/V heads (bf16 [B,H,S,D]) + context (bf16 [B,S,H,D]).
  const size_t NHD = (size_t)B_ * H_ * S_ * D_;      // 4,194,304 elems
  bf16_t* qh  = (bf16_t*)d_ws;
  bf16_t* kh  = qh + NHD;
  bf16_t* vh  = kh + NHD;
  bf16_t* ctx = vh + NHD;                            // total 33.6 MB

  qkv_proj_kernel<<<dim3(BS_ / 128, E_ / 64, 3), 256, 0, stream>>>(
      q, k, v, wq_w, wq_b, wk_w, wk_b, wv_w, wv_b, qh, kh, vh);
  attn_kernel<<<dim3(S_ / 16, H_, B_), 256, 0, stream>>>(
      qh, kh, vh, dist, mask, attw, ctx);
  out_proj_kernel<<<dim3(BS_ / 128, E_ / 64, 1), 256, 0, stream>>>(
      ctx, wo_w, wo_b, out);
}
